// SelfAttention_48112223650048
// MI455X (gfx1250) — matrix operand order repaired and checked by reading, not compiled, not run on hardware
//
#include <hip/hip_runtime.h>
#include <hip/hip_bf16.h>

// ---------------------------------------------------------------------------
// Types for CDNA5 WMMA (wave32): D(16x16 f32, 8 VGPR) = A(16x32 bf16) x B(32x16 bf16) + C
// ---------------------------------------------------------------------------
typedef __attribute__((ext_vector_type(16))) __bf16 bf16x16;
typedef __attribute__((ext_vector_type(8)))  __bf16 bf16x8;
typedef __attribute__((ext_vector_type(8)))  float  f32x8;
typedef __attribute__((ext_vector_type(4)))  int    i32x4;

union FragU {
    bf16x16 v;
    i32x4   q[2];
};

static __device__ __forceinline__ f32x8 wmma_bf16(bf16x16 a, bf16x16 b, f32x8 c) {
    // (neg_a, A, neg_b, B, c_mod, C, reuse_a, reuse_b) -> v_wmma_f32_16x16x32_bf16
    return __builtin_amdgcn_wmma_f32_16x16x32_bf16(false, a, false, b, (short)0, c, false, false);
}

// Two 16-byte chunks (8 bf16 each) -> this lane's slice of a 16x32 fragment.
static __device__ __forceinline__ bf16x16 ld_pair(const __bf16* p0, const __bf16* p1) {
    FragU f;
    f.q[0] = *reinterpret_cast<const i32x4*>(p0);
    f.q[1] = *reinterpret_cast<const i32x4*>(p1);
    return f.v;
}

// A-matrix fragment (16x32 bf16, row-major source, row = lane&15):
//   lanes 0-15:  K = k0+0..15
//   lanes 16-31: K = k0+16..31
static __device__ __forceinline__ bf16x16 ld_frag_a(const __bf16* row, int k0, int lane) {
    const __bf16* p = row + k0 + ((lane >> 4) << 4);
    return ld_pair(p, p + 8);
}

// B-matrix fragment: per-lane 16 consecutive K elements
//   lanes 0-15: K = k0..k0+15 (col = lane), lanes 16-31: K = k0+16..k0+31 (col = lane-16)
static __device__ __forceinline__ bf16x16 ld_frag_b16(const __bf16* p) {
    return ld_pair(p, p + 8);
}

// CDNA5 async global->LDS copy (16 bytes), tracked by ASYNCcnt.
static __device__ __forceinline__ void async_cp16(unsigned lds_addr, unsigned long long gaddr) {
    asm volatile("global_load_async_to_lds_b128 %0, %1, off"
                 :: "v"(lds_addr), "v"(gaddr) : "memory");
}

// ---------------------------------------------------------------------------
// Pure-VALU DPP16 row reductions (16-lane rows; lanes 0-15 / 16-31 independent)
// Sequence: xor1, xor2, then half-mirror / mirror (valid once quads/halves are
// uniform) -> all 16 lanes hold the row result. No LDS, no dscnt waits.
// ---------------------------------------------------------------------------
template <int CTRL>
static __device__ __forceinline__ float dpp_movf(float x) {
    int xi = __builtin_bit_cast(int, x);
    int r = __builtin_amdgcn_update_dpp(xi, xi, CTRL, 0xF, 0xF, true);
    return __builtin_bit_cast(float, r);
}

static __device__ __forceinline__ float red16_max(float x) {
    x = fmaxf(x, dpp_movf<0xB1>(x));    // quad_perm [1,0,3,2]
    x = fmaxf(x, dpp_movf<0x4E>(x));    // quad_perm [2,3,0,1]
    x = fmaxf(x, dpp_movf<0x141>(x));   // row_half_mirror
    x = fmaxf(x, dpp_movf<0x140>(x));   // row_mirror
    return x;
}

static __device__ __forceinline__ float red16_sum(float x) {
    x += dpp_movf<0xB1>(x);
    x += dpp_movf<0x4E>(x);
    x += dpp_movf<0x141>(x);
    x += dpp_movf<0x140>(x);
    return x;
}

// ---------------------------------------------------------------------------
// fp32 -> bf16 elementwise convert
// ---------------------------------------------------------------------------
__global__ void cvt_f32_bf16_kernel(const float* __restrict__ in, __bf16* __restrict__ out, size_t n) {
    size_t i = (size_t)blockIdx.x * blockDim.x + threadIdx.x;
    size_t stride = (size_t)gridDim.x * blockDim.x;
    for (; i < n; i += stride) out[i] = (__bf16)in[i];
}

// fp32 [K][N] row-major  ->  bf16 [N][K] (transposed) so GEMM B-fragments are contiguous in K
__global__ void cvt_transpose_kernel(const float* __restrict__ in, __bf16* __restrict__ out,
                                     int K, int N) {
    size_t total = (size_t)K * N;
    size_t i = (size_t)blockIdx.x * blockDim.x + threadIdx.x;
    size_t stride = (size_t)gridDim.x * blockDim.x;
    for (; i < total; i += stride) {
        int k = (int)(i / N);
        int n = (int)(i % N);
        out[(size_t)n * K + k] = (__bf16)in[i];
    }
}

// ---------------------------------------------------------------------------
// NT GEMM: C[M,N] = A[M,K] (bf16 row-major) x Bt[N,K] (bf16, B pre-transposed)
// Block: 256 threads = 8 waves arranged 2(M) x 4(N); wave tile 32(M) x 64(N)
// -> 8 WMMA per 32-K step per wave with B-fragments reused across two A-frags.
// Block tile: 64 x 256.
// ---------------------------------------------------------------------------
template <typename OutT>
__global__ __launch_bounds__(256) void gemm_nt_bf16_kernel(
        const __bf16* __restrict__ A, const __bf16* __restrict__ Bt,
        OutT* __restrict__ C, int M, int N, int K) {
    const int lane = threadIdx.x & 31;
    const int wave = threadIdx.x >> 5;
    const int wm = wave & 1;
    const int wn = wave >> 1;
    const int mBase = blockIdx.x * 64 + wm * 32;
    const int nBase = blockIdx.y * 256 + wn * 64;

    f32x8 acc[2][4];
#pragma unroll
    for (int mi = 0; mi < 2; ++mi)
#pragma unroll
        for (int t = 0; t < 4; ++t) acc[mi][t] = f32x8{};

    const __bf16* arow0 = A + (size_t)(mBase + (lane & 15)) * K;
    const __bf16* arow1 = arow0 + (size_t)16 * K;
    const __bf16* brow[4];
#pragma unroll
    for (int t = 0; t < 4; ++t)
        brow[t] = Bt + (size_t)(nBase + t * 16 + (lane & 15)) * K + ((lane >> 4) << 4);

    for (int k0 = 0; k0 < K; k0 += 32) {
        __builtin_prefetch(arow0 + k0 + 512, 0, 1);   // global_prefetch next K slab
        bf16x16 a0 = ld_frag_a(arow0, k0, lane);
        bf16x16 a1 = ld_frag_a(arow1, k0, lane);
#pragma unroll
        for (int t = 0; t < 4; ++t) {
            bf16x16 b = ld_frag_b16(brow[t] + k0);
            acc[0][t] = wmma_bf16(a0, b, acc[0][t]);
            acc[1][t] = wmma_bf16(a1, b, acc[1][t]);
        }
    }

    // D layout: VGPR r, lane n -> row base + r + 8*(n>>4), col nBase + t*16 + (n&15)
    const int col = lane & 15;
#pragma unroll
    for (int mi = 0; mi < 2; ++mi) {
        const int rowLo = mBase + mi * 16 + ((lane >> 4) << 3);
#pragma unroll
        for (int t = 0; t < 4; ++t) {
#pragma unroll
            for (int r = 0; r < 8; ++r) {
                C[(size_t)(rowLo + r) * N + nBase + t * 16 + col] = (OutT)acc[mi][t][r];
            }
        }
    }
}

// ---------------------------------------------------------------------------
// RoPE + QKV split.
// qkv bf16 [B,T,3C] ->
//   qr, kr : bf16 [B, nh, T, hd] (RoPE applied)
//   vt     : bf16 [B, nh, hd, T] (transposed so PV B-fragments are contiguous)
// ---------------------------------------------------------------------------
__global__ void rope_split_kernel(const __bf16* __restrict__ qkv,
                                  __bf16* __restrict__ qr, __bf16* __restrict__ kr,
                                  __bf16* __restrict__ vt, int B, int T) {
    constexpr int nh = 16, hd = 128, C = 2048;
    size_t total = (size_t)B * T * nh * (hd / 2);
    size_t idx = (size_t)blockIdx.x * blockDim.x + threadIdx.x;
    if (idx >= total) return;

    const int j  = (int)(idx & 63);          // pair index: dims (2j, 2j+1)
    const int h  = (int)((idx >> 6) & 15);
    const size_t bt = idx >> 10;
    const int t  = (int)(bt % T);
    const int b  = (int)(bt / T);

    const __bf16* row = qkv + (size_t)(b * T + t) * (3 * C);
    const int d0 = 2 * j, d1 = 2 * j + 1;

    float inv_freq = __powf(10000.0f, -((float)(2 * j) / (float)hd));
    float ang = (float)t * inv_freq;
    float s, c;
    __sincosf(ang, &s, &c);

    float q0 = (float)row[h * hd + d0],         q1 = (float)row[h * hd + d1];
    float k0 = (float)row[C + h * hd + d0],     k1 = (float)row[C + h * hd + d1];
    float v0 = (float)row[2 * C + h * hd + d0], v1 = (float)row[2 * C + h * hd + d1];

    const size_t bh = (size_t)b * nh + h;
    __bf16* qo = qr + (bh * T + t) * hd;
    qo[d0] = (__bf16)(q0 * c - q1 * s);
    qo[d1] = (__bf16)(q1 * c + q0 * s);
    __bf16* ko = kr + (bh * T + t) * hd;
    ko[d0] = (__bf16)(k0 * c - k1 * s);
    ko[d1] = (__bf16)(k1 * c + k0 * s);
    __bf16* vo = vt + bh * (size_t)hd * T;
    vo[(size_t)d0 * T + t] = (__bf16)v0;
    vo[(size_t)d1 * T + t] = (__bf16)v1;
}

// ---------------------------------------------------------------------------
// Flash attention (no mask). One wave per (b,h, 16-query tile), 4 waves/block;
// all 4 waves share (b,h), so K/V 32-key blocks are staged once per block into
// LDS via async global->LDS b128 copies (ASYNCcnt), double-buffered so the
// next block's DMA overlaps the current block's 16 WMMAs + online softmax.
// Softmax row reductions are pure-VALU DPP16 butterflies (no LDS waits).
// ---------------------------------------------------------------------------
__global__ __launch_bounds__(128) void flash_attn_kernel(
        const __bf16* __restrict__ qr, const __bf16* __restrict__ kr,
        const __bf16* __restrict__ vt, __bf16* __restrict__ attn, int B, int T) {
    constexpr int nh = 16, hd = 128, C = 2048;
    const int lane = threadIdx.x & 31;
    const int wave = threadIdx.x >> 5;
    const int tid  = threadIdx.x;

    __shared__ __bf16 kbuf[2][32 * 128];      // 2 x 8 KB : K block [32 keys][hd]
    __shared__ __bf16 vbuf[2][128 * 32];      // 2 x 8 KB : V^T block [hd][32 keys]
    __shared__ __bf16 pbuf_all[4][16 * 32];   // 1 KB per wave: P bounce

    __bf16* pb = pbuf_all[wave];

    const int qtiles = T / 16;                // 128: divisible by 4 -> waves share bh
    const int gw = blockIdx.x * 4 + wave;
    const int qt = gw % qtiles;
    const int bh = gw / qtiles;

    const __bf16* qbase = qr + (size_t)bh * T * hd;
    const unsigned long long kg0 = (unsigned long long)(uintptr_t)(kr + (size_t)bh * T * hd);
    const unsigned long long vg0 = (unsigned long long)(uintptr_t)(vt + (size_t)bh * hd * T);

    // ---- async stage of one 32-key block into buffer `buf` (8 x b128 per thread)
    auto stage = [&](int kb, int buf) {
        unsigned kl = (unsigned)(uintptr_t)(&kbuf[buf][0]);
        unsigned vl = (unsigned)(uintptr_t)(&vbuf[buf][0]);
        unsigned long long kg = kg0 + (unsigned long long)kb * (hd * 2);
#pragma unroll
        for (int c2 = 0; c2 < 4; ++c2)
            async_cp16(kl + c2 * 2048 + tid * 16, kg + c2 * 2048 + tid * 16);
        unsigned long long vg = vg0 + ((unsigned long long)tid * T + kb) * 2;
#pragma unroll
        for (int c2 = 0; c2 < 4; ++c2)
            async_cp16(vl + tid * 64 + c2 * 16, vg + c2 * 16);
    };

    stage(0, 0);   // prologue DMA overlaps Q fragment loads

    // Q fragments: 4 chunks of 16x32 over hd=128
    bf16x16 qa[4];
    const __bf16* qrow = qbase + (size_t)(qt * 16 + (lane & 15)) * hd;
#pragma unroll
    for (int cchunk = 0; cchunk < 4; ++cchunk)
        qa[cchunk] = ld_frag_a(qrow, cchunk * 32, lane);

    f32x8 o[8];
#pragma unroll
    for (int t = 0; t < 8; ++t) o[t] = f32x8{};
    float mrow[8], lrow[8];
#pragma unroll
    for (int r = 0; r < 8; ++r) { mrow[r] = -3.0e38f; lrow[r] = 0.0f; }

    const float scale = 0.08838834764831845f;  // 1/sqrt(128)
    const int khalf = (lane >> 4) << 4;        // +16 K elems for upper lane half
    const int nb = T / 32;

    for (int i = 0; i < nb; ++i) {
        if (i + 1 < nb) {
            stage((i + 1) * 32, (i + 1) & 1);
            asm volatile("s_wait_asynccnt 0x8" ::: "memory");   // block i landed
        } else {
            asm volatile("s_wait_asynccnt 0x0" ::: "memory");
        }
        __syncthreads();

        const __bf16* kb_s = kbuf[i & 1];
        const __bf16* vb_s = vbuf[i & 1];

        // ---- scores: two 16x16 tiles (keys 0-15, 16-31 of this block)
        f32x8 s0 = f32x8{}, s1 = f32x8{};
#pragma unroll
        for (int cchunk = 0; cchunk < 4; ++cchunk) {
            const __bf16* k0p = kb_s + (lane & 15) * hd + cchunk * 32 + khalf;
            const __bf16* k1p = k0p + 16 * hd;
            s0 = wmma_bf16(qa[cchunk], ld_frag_b16(k0p), s0);
            s1 = wmma_bf16(qa[cchunk], ld_frag_b16(k1p), s1);
        }

        // ---- online softmax (rows live across each 16-lane half)
#pragma unroll
        for (int r = 0; r < 8; ++r) {
            float v0 = s0[r] * scale;
            float v1 = s1[r] * scale;
            float mx = red16_max(fmaxf(v0, v1));
            float mnew = fmaxf(mrow[r], mx);
            float alpha = __expf(mrow[r] - mnew);
            float p0 = __expf(v0 - mnew);
            float p1 = __expf(v1 - mnew);
            float rs = red16_sum(p0 + p1);
            lrow[r] = lrow[r] * alpha + rs;
            mrow[r] = mnew;
            s0[r] = p0;
            s1[r] = p1;
#pragma unroll
            for (int t = 0; t < 8; ++t) o[t][r] *= alpha;
        }

        // ---- P relayout: C-layout -> LDS (bf16 [16 rows][32 keys]) -> A-layout
        {
            const int rowBase = (lane >> 4) << 3;
            const int col = lane & 15;
#pragma unroll
            for (int r = 0; r < 8; ++r) {
                pb[(rowBase + r) * 32 + col]      = (__bf16)s0[r];
                pb[(rowBase + r) * 32 + 16 + col] = (__bf16)s1[r];
            }
        }
        const __bf16* prow = pb + (lane & 15) * 32 + ((lane >> 4) << 4);
        bf16x16 pa = ld_pair(prow, prow + 8);

        // ---- O += P V  (8 hd-tiles of 16), V^T rows contiguous in keys
#pragma unroll
        for (int t = 0; t < 8; ++t) {
            const __bf16* vp = vb_s + (t * 16 + (lane & 15)) * 32 + khalf;
            o[t] = wmma_bf16(pa, ld_frag_b16(vp), o[t]);
        }

        __syncthreads();   // safe to overwrite buf[i&1] next iteration
    }

    // ---- epilogue: normalize and scatter to attn [B,T,C] (col = h*hd + d), bf16
    const int b = bh / nh;
    const int h = bh % nh;
    const int qrow0 = qt * 16 + ((lane >> 4) << 3);
    const int col = lane & 15;
#pragma unroll
    for (int r = 0; r < 8; ++r) {
        float invl = 1.0f / lrow[r];
        __bf16* orow = attn + (size_t)(b * T + qrow0 + r) * C + h * hd;
#pragma unroll
        for (int t = 0; t < 8; ++t)
            orow[t * 16 + col] = (__bf16)(o[t][r] * invl);
    }
}

// ---------------------------------------------------------------------------
// Host launch
// ---------------------------------------------------------------------------
extern "C" void kernel_launch(void* const* d_in, const int* in_sizes, int n_in,
                              void* d_out, int out_size, void* d_ws, size_t ws_size,
                              hipStream_t stream) {
    (void)in_sizes; (void)n_in; (void)out_size; (void)ws_size;
    constexpr int B = 4, T = 2048, C = 2048, N3 = 3 * C;
    constexpr size_t M = (size_t)B * T;               // 8192

    const float* x      = (const float*)d_in[0];
    const float* w_qkv  = (const float*)d_in[1];
    const float* w_proj = (const float*)d_in[2];
    float* out = (float*)d_out;

    char* ws = (char*)d_ws;
    // workspace layout (all 256-aligned); attn aliases xb (dead after GEMM1)
    __bf16* xb     = (__bf16*)(ws + 0);                       //  32 MB  [M, C]
    __bf16* wqkvT  = (__bf16*)(ws + 33554432);                //  24 MB  [3C, C]
    __bf16* wprojT = (__bf16*)(ws + 58720256);                //   8 MB  [C, C]
    __bf16* qkv    = (__bf16*)(ws + 67108864);                //  96 MB  [M, 3C]
    __bf16* qr     = (__bf16*)(ws + 167772160);               //  32 MB  [B,nh,T,hd]
    __bf16* kr     = (__bf16*)(ws + 201326592);               //  32 MB  [B,nh,T,hd]
    __bf16* vt     = (__bf16*)(ws + 234881024);               //  32 MB  [B,nh,hd,T]
    __bf16* attn   = xb;                                      //  32 MB  [M, C]

    // 1) converts / transposes
    cvt_f32_bf16_kernel<<<4096, 256, 0, stream>>>(x, xb, M * C);
    cvt_transpose_kernel<<<4096, 256, 0, stream>>>(w_qkv, wqkvT, C, N3);
    cvt_transpose_kernel<<<4096, 256, 0, stream>>>(w_proj, wprojT, C, C);

    // 2) QKV GEMM: [M, C] x [C, 3C] -> bf16 [M, 3C]
    gemm_nt_bf16_kernel<__bf16><<<dim3(M / 64, N3 / 256), 256, 0, stream>>>(
        xb, wqkvT, qkv, (int)M, N3, C);

    // 3) RoPE + split + V transpose
    {
        size_t total = M * 16 * 64;
        rope_split_kernel<<<(unsigned)((total + 255) / 256), 256, 0, stream>>>(
            qkv, qr, kr, vt, B, T);
    }

    // 4) flash attention: one wave per (b,h,16-query tile)
    {
        int waves = B * 16 * (T / 16);    // 8192
        flash_attn_kernel<<<waves / 4, 128, 0, stream>>>(qr, kr, vt, attn, B, T);
    }

    // 5) output projection: [M, C] x [C, C] -> fp32 d_out
    gemm_nt_bf16_kernel<float><<<dim3(M / 64, C / 256), 256, 0, stream>>>(
        attn, wprojT, out, (int)M, C, C);
}
